// Discriminator_51814485459347
// MI455X (gfx1250) — matrix operand order repaired and checked by reading, not compiled, not run on hardware
//
#include <hip/hip_runtime.h>
#include <hip/hip_bf16.h>

// ---------------------------------------------------------------------------
// GRU discriminator, reduced to the single-sequence computation implied by the
// reference's batch-broadcast bug.
//   VOCAB=32000, H=1024, L=256, B=64
// d_out = [feat: 64*256*1024 f32][out: 64 f32]
// ---------------------------------------------------------------------------

typedef __attribute__((ext_vector_type(16))) _Float16 v16h;
typedef __attribute__((ext_vector_type(8)))  _Float16 v8h;
typedef __attribute__((ext_vector_type(8)))  float    v8f;

#define HDIM   1024
#define H3     3072
#define LSEQ   256
#define BATCH  64
#define VOCABN 32000

// ------------------------- CDNA5 async copy: global -> LDS (ASYNCcnt path)
__device__ __forceinline__ void async_load_b128(unsigned lds_byte_addr,
                                                const void* gaddr) {
  asm volatile("global_load_async_to_lds_b128 %0, %1, off"
               :: "v"(lds_byte_addr), "v"(gaddr)
               : "memory");
}
__device__ __forceinline__ void wait_asynccnt0() {
  asm volatile("s_wait_asynccnt 0x0" ::: "memory");
}

// ---------------------------------------------------------------- f32 -> f16
__global__ __launch_bounds__(256) void f32_to_f16(const float* __restrict__ src,
                                                  _Float16* __restrict__ dst, int n) {
  int i = blockIdx.x * blockDim.x + threadIdx.x;
  int stride = gridDim.x * blockDim.x;
  for (; i < n; i += stride) dst[i] = (_Float16)src[i];
}

// -------------------------------------------- gather emb rows for batch 0
__global__ __launch_bounds__(256) void gather_embed(const int* __restrict__ x,
                                                    const float* __restrict__ emb,
                                                    _Float16* __restrict__ e0) {
  int l = blockIdx.x;                 // 0..255
  int idx = x[l * BATCH];             // batch element 0
  idx = idx < 0 ? 0 : (idx >= VOCABN ? VOCABN - 1 : idx);
  const float* src = emb + (long long)idx * HDIM;
  for (int k = threadIdx.x; k < HDIM; k += blockDim.x)
    e0[l * HDIM + k] = (_Float16)src[k];
}

// ------------------------------------------------------------------- zero h
__global__ void zero_buf(float* __restrict__ p, int n) {
  int i = blockIdx.x * blockDim.x + threadIdx.x;
  if (i < n) p[i] = 0.f;
}

// ---------------------------------------------------------------------------
// gi = e0 @ w_ih^T + b_ih   :  M=256, N=3072, K=1024, f16 in / f32 out.
// Block = 256 threads (8 waves). Block tile: M=16, N=256; wave tile 16x32
// (two 16x16 WMMA accumulators sharing one A fragment).
// A tile (16x1024 f16 = 32KB) is staged into LDS once via the CDNA5 async
// copy engine (global_load_async_to_lds_b128 / s_wait_asynccnt), then A
// fragments are ds_load'ed; B streams from global (L2-resident w_ih).
// Fragment k order: contiguous, k = 16*half + s for both A and B (wave32).
// ---------------------------------------------------------------------------
__global__ __launch_bounds__(256) void gemm_gi(const _Float16* __restrict__ A,   // [256][1024]
                                               const _Float16* __restrict__ Bw,  // [3072][1024]
                                               const float* __restrict__ bias,   // [3072]
                                               float* __restrict__ C) {          // [256][3072]
  __shared__ _Float16 Atile[16 * HDIM];           // 32 KB

  const int lane = threadIdx.x & 31;
  const int wave = threadIdx.x >> 5;              // 0..7
  const int l15  = lane & 15;
  const int half = lane >> 4;                     // 0 or 1
  const int m0   = blockIdx.y * 16;
  const int n0   = blockIdx.x * 256 + wave * 32;  // two 16-wide subtiles

  // ---- async-stage the contiguous 32KB A block (rows m0..m0+15) into LDS
  {
    const char* gsrc = (const char*)(A + (size_t)m0 * HDIM);
    unsigned ldsbase = (unsigned)(uintptr_t)(&Atile[0]);
    for (int t = threadIdx.x; t < 2048; t += 256)   // 2048 x 16B = 32KB
      async_load_b128(ldsbase + (unsigned)t * 16u, gsrc + (size_t)t * 16u);
    wait_asynccnt0();
  }
  __syncthreads();

  const _Float16* arow  = Atile + l15 * HDIM + half * 16;             // LDS
  const _Float16* bcol0 = Bw + (size_t)(n0 + l15) * HDIM + half * 16; // global
  const _Float16* bcol1 = bcol0 + (size_t)16 * HDIM;

  v8f c0 = {}, c1 = {};
  for (int k0 = 0; k0 < HDIM; k0 += 32) {
    v8h a0 = *(const v8h*)(arow + k0);        // ds_load_b128
    v8h a1 = *(const v8h*)(arow + k0 + 8);
    v16h a = __builtin_shufflevector(a0, a1, 0,1,2,3,4,5,6,7,8,9,10,11,12,13,14,15);
    v16h b0 = *(const v16h*)(bcol0 + k0);     // global 32B
    v16h b1 = *(const v16h*)(bcol1 + k0);
    c0 = __builtin_amdgcn_wmma_f32_16x16x32_f16(false, a, false, b0, (short)0, c0, false, false);
    c1 = __builtin_amdgcn_wmma_f32_16x16x32_f16(false, a, false, b1, (short)0, c1, false, false);
  }

  const int row = m0 + half * 8;
  const int col0 = n0 + l15;
  const float bv0 = bias[col0];
  const float bv1 = bias[col0 + 16];
#pragma unroll
  for (int r = 0; r < 8; ++r) {
    C[(size_t)(row + r) * H3 + col0]      = c0[r] + bv0;
    C[(size_t)(row + r) * H3 + col0 + 16] = c1[r] + bv1;
  }
}

// ---------------------------------------------------------------------------
// One GRU step (single sequence). gh = h @ w_hh^T for 3 gate rows per unit,
// lane-parallel over K with wave32 shuffle reduction; h staged in LDS.
// ---------------------------------------------------------------------------
__device__ __forceinline__ float sigmoidf_(float x) { return 1.f / (1.f + __expf(-x)); }

__global__ __launch_bounds__(256) void gru_step(const float* __restrict__ gi_l,   // [3072]
                                                const _Float16* __restrict__ whh, // [3072][1024] f16
                                                const float* __restrict__ bhh,    // [3072]
                                                const float* __restrict__ h_in,   // [1024]
                                                float* __restrict__ h_out,        // [1024]
                                                float* __restrict__ hs_l) {       // [1024]
  __shared__ float hsm[HDIM];
  for (int k = threadIdx.x; k < HDIM; k += 256) hsm[k] = h_in[k];
  __syncthreads();

  const int wave = threadIdx.x >> 5;
  const int lane = threadIdx.x & 31;
  const int jbase = blockIdx.x * 32 + wave * 4;   // 32 blocks * 32 units = 1024
  const int kb = lane * 32;                       // 32 contiguous K per lane

  for (int u = 0; u < 4; ++u) {
    const int j = jbase + u;
    const _Float16* wr = whh + (size_t)j * HDIM;
    const _Float16* wz = wr + (size_t)HDIM * HDIM;
    const _Float16* wn = wr + (size_t)2 * HDIM * HDIM;

    if (u < 3) {  // prefetch next unit's gate rows into cache
      __builtin_prefetch(wr + HDIM + kb, 0, 1);
      __builtin_prefetch(wz + HDIM + kb, 0, 1);
      __builtin_prefetch(wn + HDIM + kb, 0, 1);
    }

    float accr = 0.f, accz = 0.f, accn = 0.f;
#pragma unroll
    for (int c = 0; c < 4; ++c) {
      v8h r8 = *(const v8h*)(wr + kb + c * 8);
      v8h z8 = *(const v8h*)(wz + kb + c * 8);
      v8h n8 = *(const v8h*)(wn + kb + c * 8);
#pragma unroll
      for (int e = 0; e < 8; ++e) {
        float hv = hsm[kb + c * 8 + e];
        accr += (float)r8[e] * hv;
        accz += (float)z8[e] * hv;
        accn += (float)n8[e] * hv;
      }
    }
    // wave32 tree reduction
    for (int off = 16; off > 0; off >>= 1) {
      accr += __shfl_down(accr, off);
      accz += __shfl_down(accz, off);
      accn += __shfl_down(accn, off);
    }
    if (lane == 0) {
      float r  = sigmoidf_(gi_l[j]            + accr + bhh[j]);
      float z  = sigmoidf_(gi_l[j + HDIM]     + accz + bhh[j + HDIM]);
      float hn = accn + bhh[j + 2 * HDIM];
      float n  = tanhf(gi_l[j + 2 * HDIM] + r * hn);
      float h2 = (1.f - z) * n + z * hsm[j];
      h_out[j] = h2;
      hs_l[j]  = h2;
    }
  }
}

// ------------------------------------- feat = broadcast hs across B (float4)
__global__ __launch_bounds__(256) void broadcast_feat(const float4* __restrict__ hs4,
                                                      float4* __restrict__ out4) {
  const size_t total4 = (size_t)BATCH * (LSEQ * HDIM / 4);   // 64 * 65536
  size_t i = (size_t)blockIdx.x * blockDim.x + threadIdx.x;
  size_t stride = (size_t)gridDim.x * blockDim.x;
  for (; i < total4; i += stride)
    out4[i] = hs4[i & ((LSEQ * HDIM / 4) - 1)];              // mask 65535
}

// -------------------- out[b] = sigmoid(dot(hs, w_out) + b_out), b = 0..63
__global__ __launch_bounds__(1024) void out_head(const float* __restrict__ hs,
                                                 const float* __restrict__ wout,
                                                 const float* __restrict__ bout,
                                                 float* __restrict__ out) {
  __shared__ float red[32];
  float acc = 0.f;
  for (int i = threadIdx.x; i < LSEQ * HDIM; i += 1024)
    acc += hs[i] * wout[i];
  for (int off = 16; off > 0; off >>= 1) acc += __shfl_down(acc, off);
  const int wave = threadIdx.x >> 5, lane = threadIdx.x & 31;
  if (lane == 0) red[wave] = acc;
  __syncthreads();
  if (wave == 0) {
    float v = red[lane];
    for (int off = 16; off > 0; off >>= 1) v += __shfl_down(v, off);
    if (lane == 0) red[0] = sigmoidf_(v + bout[0]);
  }
  __syncthreads();
  if (threadIdx.x < BATCH) out[threadIdx.x] = red[0];
}

// ---------------------------------------------------------------------------
extern "C" void kernel_launch(void* const* d_in, const int* in_sizes, int n_in,
                              void* d_out, int out_size, void* d_ws, size_t ws_size,
                              hipStream_t stream) {
  (void)in_sizes; (void)n_in; (void)out_size; (void)ws_size;

  const int*   x     = (const int*)d_in[0];
  const float* emb   = (const float*)d_in[1];
  const float* w_ih  = (const float*)d_in[2];
  const float* w_hh  = (const float*)d_in[3];
  const float* b_ih  = (const float*)d_in[4];
  const float* b_hh  = (const float*)d_in[5];
  const float* w_out = (const float*)d_in[6];
  const float* b_out = (const float*)d_in[7];

  char* ws = (char*)d_ws;
  _Float16* e0    = (_Float16*)ws; ws += (size_t)LSEQ * HDIM * 2;  // 512 KB
  _Float16* wih16 = (_Float16*)ws; ws += (size_t)H3 * HDIM * 2;    // 6 MB
  _Float16* whh16 = (_Float16*)ws; ws += (size_t)H3 * HDIM * 2;    // 6 MB
  float*    gi    = (float*)ws;    ws += (size_t)LSEQ * H3 * 4;    // 3 MB
  float*    hsbuf = (float*)ws;    ws += (size_t)LSEQ * HDIM * 4;  // 1 MB
  float*    hping = (float*)ws;    ws += 2 * HDIM * 4;             // 8 KB

  // precompute: f16 weights, batch-0 embeddings, h0 = 0
  f32_to_f16<<<768, 256, 0, stream>>>(w_ih, wih16, H3 * HDIM);
  f32_to_f16<<<768, 256, 0, stream>>>(w_hh, whh16, H3 * HDIM);
  gather_embed<<<LSEQ, 256, 0, stream>>>(x, emb, e0);
  zero_buf<<<8, 256, 0, stream>>>(hping, 2 * HDIM);

  // gi = e0 @ w_ih^T + b_ih  (WMMA, async-LDS staged A)
  gemm_gi<<<dim3(H3 / 256, LSEQ / 16), 256, 0, stream>>>(e0, wih16, b_ih, gi);

  // sequential scan, ping-pong h
  for (int l = 0; l < LSEQ; ++l) {
    const float* hin  = hping + (l & 1) * HDIM;
    float*       hout = hping + ((l + 1) & 1) * HDIM;
    gru_step<<<32, 256, 0, stream>>>(gi + (size_t)l * H3, whh16, b_hh,
                                     hin, hout, hsbuf + (size_t)l * HDIM);
  }

  // outputs
  float* feat = (float*)d_out;
  broadcast_feat<<<4096, 256, 0, stream>>>((const float4*)hsbuf, (float4*)feat);
  out_head<<<1, 1024, 0, stream>>>(hsbuf, w_out, b_out,
                                   feat + (size_t)BATCH * LSEQ * HDIM);
}
